// PointNetEnc_20478404067353
// MI455X (gfx1250) — matrix operand order repaired and checked by reading, not compiled, not run on hardware
//
#include <hip/hip_runtime.h>
#include <hip/hip_bf16.h>
#include <cstdint>

// ---------------------------------------------------------------------------
// PointNet++ encoder (VAE head) for gfx1250 / MI455X.
// - MLPs on v_wmma_f32_16x16x32_f16, fp16 transposed weights (B contiguous/lane)
// - fp16 activations (x1/x2) -> halves scattered gather traffic
// - SA2/SA3 A-tile staging via GLOBAL_LOAD_ASYNC_TO_LDS_B128 when available
// ---------------------------------------------------------------------------

typedef __attribute__((ext_vector_type(16))) _Float16 v16h;
typedef __attribute__((ext_vector_type(8)))  _Float16 v8h;
typedef __attribute__((ext_vector_type(8)))  float    v8f;

static constexpr int   BB    = 8;
static constexpr int   N1    = 4096;
static constexpr int   M1    = 2048;
static constexpr int   M2    = 512;
static constexpr int   KNB   = 64;
static constexpr float INF_F = 3.4e38f;

// ---------------------------------------------------------------------------
// Async global->LDS copy (CDNA5). The builtin takes (int4-as1*, int4-as3*,
// imm offset, imm cpol) per the clang diagnostic from the previous probe.
// ---------------------------------------------------------------------------
#if defined(__gfx1250__) && __has_builtin(__builtin_amdgcn_global_load_async_to_lds_b128)
#define HAVE_ASYNC_LDS 1
typedef int vsi4 __attribute__((vector_size(16)));
typedef __attribute__((address_space(1))) vsi4 g1_vsi4;
typedef __attribute__((address_space(3))) vsi4 l3_vsi4;
__device__ __forceinline__ void async_copy16(const void* g, void* l) {
  __builtin_amdgcn_global_load_async_to_lds_b128(
      (g1_vsi4*)(unsigned long long)(uintptr_t)g,
      (l3_vsi4*)(unsigned int)(uintptr_t)l, 0, 0);
}
__device__ __forceinline__ void wait_async0() {
#if __has_builtin(__builtin_amdgcn_s_wait_asynccnt)
  __builtin_amdgcn_s_wait_asynccnt(0);
#else
  asm volatile("s_wait_asynccnt 0x0" ::: "memory");
#endif
  asm volatile("" ::: "memory");
}
#else
#define HAVE_ASYNC_LDS 0
#endif

// ---------------------------------------------------------------------------
// Weight conversion: fp32 W[Cin][Cout] -> fp16 WT[Cout][Kpad] (transposed,
// zero-padded K). Lane loading column N of B then reads two 8-half runs, 16 apart in K.
// ---------------------------------------------------------------------------
__global__ void wtrans_kernel(const float* __restrict__ w, _Float16* __restrict__ wt,
                              int cin, int cout, int kpad) {
  int i = blockIdx.x * blockDim.x + threadIdx.x;
  int total = cout * kpad;
  if (i >= total) return;
  int o = i / kpad, k = i % kpad;
  wt[i] = (k < cin) ? (_Float16)w[k * cout + o] : (_Float16)0.0f;
}

// ---------------------------------------------------------------------------
// WMMA fragment loaders (CDNA5 16-bit layouts, wave32).
// A 16x32: lane&15 = row; lane<16 holds K{0..7,16..23}, lane>=16 K{8..15,24..31}.
// B 32x16: lane&15 = col; lane<16 holds K{0..7,16..23}, lane>=16 K{8..15,24..31}.
// ---------------------------------------------------------------------------
__device__ __forceinline__ v16h ld_a(const _Float16* A, int as, int t, int kc, int lane) {
  const int row = t * 16 + (lane & 15);
  const int kb  = kc * 32 + ((lane & 16) ? 8 : 0);
  const _Float16* p = A + row * as + kb;
  const v8h lo = *(const v8h*)(p);
  const v8h hi = *(const v8h*)(p + 16);
  v16h r;
#pragma unroll
  for (int i = 0; i < 8; ++i) { r[i] = lo[i]; r[i + 8] = hi[i]; }
  return r;
}

__device__ __forceinline__ v16h ld_b(const _Float16* WT, int ws, int n, int kc, int lane) {
  const int col = n * 16 + (lane & 15);
  const int kb  = kc * 32 + ((lane & 16) ? 8 : 0);
  const _Float16* p = WT + col * ws + kb;
  const v8h lo = *(const v8h*)(p);
  const v8h hi = *(const v8h*)(p + 16);
  v16h r;
#pragma unroll
  for (int i = 0; i < 8; ++i) { r[i] = lo[i]; r[i + 8] = hi[i]; }
  return r;
}

// One MLP layer: H = act(A @ W + b). B fragments hoisted across row tiles.
template <bool RELU, int KCH>
__device__ __forceinline__ void mlp_layer(const _Float16* A, int as,
                                          const _Float16* WT, int ws,
                                          const float* __restrict__ bias,
                                          _Float16* H, int hs,
                                          int ntiles, int rtiles, int lane) {
  for (int n = 0; n < ntiles; ++n) {
    const int col  = n * 16 + (lane & 15);
    const float bv = bias[col];
    v16h bfr[KCH];
#pragma unroll
    for (int kc = 0; kc < KCH; ++kc) bfr[kc] = ld_b(WT, ws, n, kc, lane);
    for (int t = 0; t < rtiles; ++t) {
      v8f acc = {};
#pragma unroll
      for (int kc = 0; kc < KCH; ++kc) {
        const v16h av = ld_a(A, as, t, kc, lane);
        acc = __builtin_amdgcn_wmma_f32_16x16x32_f16(false, av, false, bfr[kc],
                                                     (short)0, acc, false, false);
      }
      const int rb = t * 16 + ((lane & 16) ? 8 : 0);
#pragma unroll
      for (int g = 0; g < 8; ++g) {
        float v = acc[g] + bv;
        if (RELU) v = v > 0.0f ? v : 0.0f;
        H[(rb + g) * hs + col] = (_Float16)v;
      }
    }
  }
}

// Final MLP layer fused with masked max over rows; lanes 0..15 write Cout cols.
template <int KCH, typename OUTT>
__device__ __forceinline__ void mlp_layer_max(const _Float16* A, int as,
                                              const _Float16* WT, int ws,
                                              const float* __restrict__ bias,
                                              const unsigned char* vf,
                                              int rtiles, int ntiles, int lane,
                                              OUTT* __restrict__ outp) {
  for (int n = 0; n < ntiles; ++n) {
    const int col  = n * 16 + (lane & 15);
    const float bv = bias[col];
    v16h bfr[KCH];
#pragma unroll
    for (int kc = 0; kc < KCH; ++kc) bfr[kc] = ld_b(WT, ws, n, kc, lane);
    float best = -INF_F;
    for (int t = 0; t < rtiles; ++t) {
      v8f acc = {};
#pragma unroll
      for (int kc = 0; kc < KCH; ++kc) {
        const v16h av = ld_a(A, as, t, kc, lane);
        acc = __builtin_amdgcn_wmma_f32_16x16x32_f16(false, av, false, bfr[kc],
                                                     (short)0, acc, false, false);
      }
      const int rb = t * 16 + ((lane & 16) ? 8 : 0);
#pragma unroll
      for (int g = 0; g < 8; ++g) {
        const bool ok = vf ? (vf[rb + g] != 0) : true;
        if (ok) {
          const float v = acc[g] + bv;
          best = best > v ? best : v;
        }
      }
    }
    const float ob = __shfl_xor(best, 16, 32);
    best = best > ob ? best : ob;
    if (lane < 16) outp[n * 16 + lane] = (OUTT)((best < -3.0e38f) ? 0.0f : best);
  }
}

// ---------------------------------------------------------------------------
// Farthest point sampling: one block per batch, deterministic start at 0.
// ---------------------------------------------------------------------------
template <int N, int M>
__global__ void fps_kernel(const float* __restrict__ pos, int* __restrict__ idx,
                           float* __restrict__ cpos) {
  __shared__ float dist[N];
  __shared__ float rv[256];
  __shared__ int   ri[256];
  __shared__ int   s_far;
  const int b = blockIdx.x;
  const int t = threadIdx.x;
  const float* p = pos + (size_t)b * N * 3;
  for (int j = t; j < N; j += 256) dist[j] = INF_F;
  if (t == 0) s_far = 0;
  __syncthreads();
  for (int m = 0; m < M; ++m) {
    const int far = s_far;
    const float fx = p[far * 3 + 0], fy = p[far * 3 + 1], fz = p[far * 3 + 2];
    if (t == 0) {
      idx[(size_t)b * M + m] = far;
      cpos[((size_t)b * M + m) * 3 + 0] = fx;
      cpos[((size_t)b * M + m) * 3 + 1] = fy;
      cpos[((size_t)b * M + m) * 3 + 2] = fz;
    }
    float bv = -1.0f; int bi = 0;
    for (int j = t; j < N; j += 256) {
      const float dx = p[j * 3 + 0] - fx;
      const float dy = p[j * 3 + 1] - fy;
      const float dz = p[j * 3 + 2] - fz;
      const float d = dx * dx + dy * dy + dz * dz;
      float dd = dist[j];
      dd = d < dd ? d : dd;
      dist[j] = dd;
      if (dd > bv) { bv = dd; bi = j; }   // strict > keeps lowest index on ties
    }
    rv[t] = bv; ri[t] = bi;
    __syncthreads();
    for (int s = 128; s > 0; s >>= 1) {
      if (t < s) {
        if (rv[t + s] > rv[t] || (rv[t + s] == rv[t] && ri[t + s] < ri[t])) {
          rv[t] = rv[t + s]; ri[t] = ri[t + s];
        }
      }
      __syncthreads();
    }
    if (t == 0) s_far = ri[0];
    __syncthreads();
  }
}

// ---------------------------------------------------------------------------
// Radius top-K: one wave per center; d2 table in LDS; 64 wave-argmin rounds
// with lowest-index tie-break (matches lax.top_k on -d2). Invalid -> -1.
// ---------------------------------------------------------------------------
template <int N, int M>
__global__ void nbr_kernel(const float* __restrict__ pos, const float* __restrict__ cent,
                           int* __restrict__ nbr, float r2) {
  constexpr int WPB = 2;
  __shared__ float d2s[WPB][N];
  const int w = threadIdx.x >> 5, lane = threadIdx.x & 31;
  const int cid = blockIdx.x * WPB + w;
  const int b = cid / M;
  const float* p = pos + (size_t)b * N * 3;
  const float cx = cent[(size_t)cid * 3 + 0];
  const float cy = cent[(size_t)cid * 3 + 1];
  const float cz = cent[(size_t)cid * 3 + 2];
  float* dd = d2s[w];
  for (int j = lane; j < N; j += 32) {
    const float dx = p[j * 3 + 0] - cx;
    const float dy = p[j * 3 + 1] - cy;
    const float dz = p[j * 3 + 2] - cz;
    const float d = dx * dx + dy * dy + dz * dz;
    dd[j] = (d <= r2) ? d : INF_F;
  }
  int* out = nbr + (size_t)cid * KNB;
  for (int k = 0; k < KNB; ++k) {
    float bv = INF_F; int bi = 0x7fffffff;
    for (int j = lane; j < N; j += 32) {
      const float v = dd[j];
      if (v < bv) { bv = v; bi = j; }
    }
#pragma unroll
    for (int s = 16; s > 0; s >>= 1) {
      const float ov = __shfl_xor(bv, s, 32);
      const int   oi = __shfl_xor(bi, s, 32);
      if (ov < bv || (ov == bv && oi < bi)) { bv = ov; bi = oi; }
    }
    if (bv >= 3.0e38f) {
      if (lane == 0) out[k] = -1;
    } else {
      if (lane == 0) out[k] = bi;
      dd[bi] = INF_F;   // same value from all lanes; LDS in-order per wave
    }
  }
}

// ---------------------------------------------------------------------------
// SA module: gather K=64 neighbor rows into LDS (f16), 3-layer WMMA MLP,
// masked max over neighbors. One wave per center, 2 waves per block.
// x_in is fp16 [NPTS][XF]; output xout fp16 [M][C3].
// ---------------------------------------------------------------------------
template <int NPTS, int M, int CIN, int KP0, int C1, int C2, int C3>
__global__ void sa_kernel(const float* __restrict__ pos, const _Float16* __restrict__ x_in,
                          const float* __restrict__ cent, const int* __restrict__ nbr,
                          const _Float16* __restrict__ wt0, const float* __restrict__ b0,
                          const _Float16* __restrict__ wt1, const float* __restrict__ b1,
                          const _Float16* __restrict__ wt2, const float* __restrict__ b2,
                          _Float16* __restrict__ xout) {
  constexpr int WPB = 2;
  constexpr int XF  = CIN - 3;
  constexpr int KP1 = C1;               // C1 is a multiple of 32
  __shared__ _Float16 A0[WPB][64 * KP0];
  __shared__ _Float16 H1[WPB][64 * KP1];
  __shared__ unsigned char vf[WPB][64];
  const int w = threadIdx.x >> 5, lane = threadIdx.x & 31;
  const int cid = blockIdx.x * WPB + w;
  const int b = cid / M;
  const int* nb = nbr + (size_t)cid * KNB;
  const float cx = cent[(size_t)cid * 3 + 0];
  const float cy = cent[(size_t)cid * 3 + 1];
  const float cz = cent[(size_t)cid * 3 + 2];
  _Float16* a0 = A0[w];
  _Float16* h1 = H1[w];

  __builtin_prefetch(wt0, 0, 3);
  __builtin_prefetch(wt2, 0, 3);

  // Gather neighbor rows: [x_j (XF f16), rel (3), zero-pad].
  for (int r = lane; r < KNB; r += 32) {
    const int j = nb[r];
    const bool valid = (j >= 0);
    vf[w][r] = valid ? 1 : 0;
    const int jj = valid ? j : 0;
    _Float16* row = a0 + r * KP0;
    if constexpr (XF > 0) {
      const _Float16* xj = x_in + ((size_t)b * NPTS + jj) * XF;
      if (valid) {
#if HAVE_ASYNC_LDS
        for (int c = 0; c < XF; c += 8) async_copy16(xj + c, row + c);
#else
        for (int c = 0; c < XF; ++c) row[c] = xj[c];
#endif
      } else {
        for (int c = 0; c < XF; ++c) row[c] = (_Float16)0.0f;
      }
    }
    const float* pj = pos + ((size_t)b * NPTS + jj) * 3;
    row[XF + 0] = valid ? (_Float16)(pj[0] - cx) : (_Float16)0.0f;
    row[XF + 1] = valid ? (_Float16)(pj[1] - cy) : (_Float16)0.0f;
    row[XF + 2] = valid ? (_Float16)(pj[2] - cz) : (_Float16)0.0f;
    for (int c = CIN; c < KP0; ++c) row[c] = (_Float16)0.0f;
  }
#if HAVE_ASYNC_LDS
  if constexpr (XF > 0) wait_async0();
#endif

  mlp_layer<true, KP0 / 32>(a0, KP0, wt0, KP0, b0, h1, KP1, C1 / 16, 4, lane);
  mlp_layer<true, KP1 / 32>(h1, KP1, wt1, KP1, b1, a0, KP0, C2 / 16, 4, lane);
  mlp_layer_max<C2 / 32, _Float16>(a0, KP0, wt2, C2, b2, vf[w], 4, C3 / 16, lane,
                                   xout + (size_t)cid * C3);
}

// ---------------------------------------------------------------------------
// SA3: per-point MLP(131->128->256->512) on concat(x2, pos2); each wave does a
// 16-point row tile, writes partial column maxima for the global max pool.
// ---------------------------------------------------------------------------
__global__ void sa3_kernel(const _Float16* __restrict__ x2, const float* __restrict__ pos2,
                           const _Float16* __restrict__ wt0, const float* __restrict__ b0,
                           const _Float16* __restrict__ wt1, const float* __restrict__ b1,
                           const _Float16* __restrict__ wt2, const float* __restrict__ b2,
                           float* __restrict__ gpart) {
  constexpr int WPB = 2;
  __shared__ _Float16 A0[WPB][16 * 160];
  __shared__ _Float16 H1[WPB][16 * 128];
  __shared__ _Float16 H2[WPB][16 * 256];
  const int w = threadIdx.x >> 5, lane = threadIdx.x & 31;
  const int tile = blockIdx.x * WPB + w;   // 0..255 (= 8 batches * 32 tiles)
  const int b = tile >> 5, tc = tile & 31;
  const int row0 = tc * 16;
  _Float16* a0 = A0[w];
  _Float16* h1 = H1[w];
  _Float16* h2 = H2[w];

  __builtin_prefetch(wt2, 0, 3);

#if HAVE_ASYNC_LDS
  // x2 rows: 16 rows x 128 halves = 16 rows x 16 chunks of 16B, async to LDS.
  for (int i = lane; i < 16 * 16; i += 32) {
    const int r = i >> 4, ch = (i & 15) * 8;
    async_copy16(x2 + ((size_t)b * M2 + row0 + r) * 128 + ch, a0 + r * 160 + ch);
  }
  for (int i = lane; i < 16 * 32; i += 32) {   // cols 128..159: pos2 + pad
    const int r = i >> 5, c = 128 + (i & 31);
    const float v = (c < 131) ? pos2[((size_t)b * M2 + row0 + r) * 3 + (c - 128)] : 0.0f;
    a0[r * 160 + c] = (_Float16)v;
  }
  wait_async0();
#else
  for (int i = lane; i < 16 * 160; i += 32) {
    const int r = i / 160, c = i % 160;
    const int pt = row0 + r;
    float v = 0.0f;
    if (c < 128)      v = (float)x2[((size_t)b * M2 + pt) * 128 + c];
    else if (c < 131) v = pos2[((size_t)b * M2 + pt) * 3 + (c - 128)];
    a0[i] = (_Float16)v;
  }
#endif

  mlp_layer<true, 5>(a0, 160, wt0, 160, b0, h1, 128, 8, 1, lane);
  mlp_layer<true, 4>(h1, 128, wt1, 128, b1, h2, 256, 16, 1, lane);
  mlp_layer_max<8, float>(h2, 256, wt2, 256, b2, nullptr, 1, 32, lane,
                          gpart + (size_t)tile * 512);
}

// Reduce the 32 per-tile partial maxima per batch into feat[b][512].
__global__ void gmax_kernel(const float* __restrict__ gpart, float* __restrict__ feat) {
  const int i = blockIdx.x * blockDim.x + threadIdx.x;
  if (i >= BB * 512) return;
  const int b = i >> 9, c = i & 511;
  float mx = -INF_F;
  for (int p = 0; p < 32; ++p) mx = fmaxf(mx, gpart[((size_t)b * 32 + p) * 512 + c]);
  feat[i] = mx;
}

// ---------------------------------------------------------------------------
// VAE head: mu/log_var projections + reparameterization (deterministic
// hash-based Gaussian noise).
// ---------------------------------------------------------------------------
__device__ __forceinline__ float hash_normal(unsigned i) {
  unsigned a = i * 747796405u + 2891336453u;
  a ^= a >> 17; a *= 0xed5ad4bbu; a ^= a >> 11; a *= 0xac4c1b51u; a ^= a >> 15;
  unsigned h = (i + 0x9e3779b9u) * 2654435761u;
  h ^= h >> 13; h *= 0x31848babu; h ^= h >> 16;
  const float u1 = (float)(a & 0xFFFFFFu) * (1.0f / 16777216.0f) + 1.0e-7f;
  const float u2 = (float)(h & 0xFFFFFFu) * (1.0f / 16777216.0f);
  return sqrtf(-2.0f * logf(u1)) * cosf(6.28318530718f * u2);
}

__global__ void head_kernel(const float* __restrict__ feat,
                            const float* __restrict__ wmu, const float* __restrict__ bmu,
                            const float* __restrict__ wlv, const float* __restrict__ blv,
                            float* __restrict__ out) {
  const int t = threadIdx.x;            // 256 = 8 batches * 32 dims
  const int b = t >> 5, o = t & 31;
  const float* f = feat + (size_t)b * 512;
  float mu = bmu[o], lv = blv[o];
  for (int c = 0; c < 512; ++c) {
    const float fv = f[c];
    mu += fv * wmu[c * 32 + o];
    lv += fv * wlv[c * 32 + o];
  }
  const float eps = hash_normal((unsigned)t);
  const float z = mu + eps * expf(0.5f * lv);
  out[t]       = z;
  out[256 + t] = mu;
  out[512 + t] = lv;
}

// ---------------------------------------------------------------------------
// Host-side orchestration.
// ---------------------------------------------------------------------------
extern "C" void kernel_launch(void* const* d_in, const int* in_sizes, int n_in,
                              void* d_out, int out_size, void* d_ws, size_t ws_size,
                              hipStream_t stream) {
  (void)in_sizes; (void)n_in; (void)out_size; (void)ws_size;
  const float* pos   = (const float*)d_in[0];
  const float* s1w0  = (const float*)d_in[1];  const float* s1b0 = (const float*)d_in[2];
  const float* s1w1  = (const float*)d_in[3];  const float* s1b1 = (const float*)d_in[4];
  const float* s1w2  = (const float*)d_in[5];  const float* s1b2 = (const float*)d_in[6];
  const float* s2w0  = (const float*)d_in[7];  const float* s2b0 = (const float*)d_in[8];
  const float* s2w1  = (const float*)d_in[9];  const float* s2b1 = (const float*)d_in[10];
  const float* s2w2  = (const float*)d_in[11]; const float* s2b2 = (const float*)d_in[12];
  const float* s3w0  = (const float*)d_in[13]; const float* s3b0 = (const float*)d_in[14];
  const float* s3w1  = (const float*)d_in[15]; const float* s3b1 = (const float*)d_in[16];
  const float* s3w2  = (const float*)d_in[17]; const float* s3b2 = (const float*)d_in[18];
  const float* wmu   = (const float*)d_in[19]; const float* bmu  = (const float*)d_in[20];
  const float* wlv   = (const float*)d_in[21]; const float* blv  = (const float*)d_in[22];

  char* ws = (char*)d_ws;
  size_t off = 0;
  auto alloc = [&](size_t bytes) -> char* {
    char* p = ws + off;
    off += (bytes + 255) & ~(size_t)255;
    return p;
  };

  _Float16* wt10 = (_Float16*)alloc((size_t)32  * 32  * 2);
  _Float16* wt11 = (_Float16*)alloc((size_t)32  * 32  * 2);
  _Float16* wt12 = (_Float16*)alloc((size_t)64  * 32  * 2);
  _Float16* wt20 = (_Float16*)alloc((size_t)64  * 96  * 2);
  _Float16* wt21 = (_Float16*)alloc((size_t)64  * 64  * 2);
  _Float16* wt22 = (_Float16*)alloc((size_t)128 * 64  * 2);
  _Float16* wt30 = (_Float16*)alloc((size_t)128 * 160 * 2);
  _Float16* wt31 = (_Float16*)alloc((size_t)256 * 128 * 2);
  _Float16* wt32 = (_Float16*)alloc((size_t)512 * 256 * 2);
  int*      idx1  = (int*)     alloc((size_t)BB * M1 * 4);
  float*    pos1  = (float*)   alloc((size_t)BB * M1 * 3 * 4);
  int*      nbr1  = (int*)     alloc((size_t)BB * M1 * KNB * 4);
  _Float16* x1    = (_Float16*)alloc((size_t)BB * M1 * 64 * 2);
  int*      idx2  = (int*)     alloc((size_t)BB * M2 * 4);
  float*    pos2  = (float*)   alloc((size_t)BB * M2 * 3 * 4);
  int*      nbr2  = (int*)     alloc((size_t)BB * M2 * KNB * 4);
  _Float16* x2    = (_Float16*)alloc((size_t)BB * M2 * 128 * 2);
  float*    gpart = (float*)   alloc((size_t)BB * 32 * 512 * 4);
  float*    feat  = (float*)   alloc((size_t)BB * 512 * 4);

  auto wt_launch = [&](const float* w, _Float16* wt, int cin, int cout, int kpad) {
    const int total = cout * kpad;
    wtrans_kernel<<<(total + 255) / 256, 256, 0, stream>>>(w, wt, cin, cout, kpad);
  };
  wt_launch(s1w0, wt10, 3,   32,  32);
  wt_launch(s1w1, wt11, 32,  32,  32);
  wt_launch(s1w2, wt12, 32,  64,  32);
  wt_launch(s2w0, wt20, 67,  64,  96);
  wt_launch(s2w1, wt21, 64,  64,  64);
  wt_launch(s2w2, wt22, 64,  128, 64);
  wt_launch(s3w0, wt30, 131, 128, 160);
  wt_launch(s3w1, wt31, 128, 256, 128);
  wt_launch(s3w2, wt32, 256, 512, 256);

  // ---- SA1 ----
  fps_kernel<N1, M1><<<BB, 256, 0, stream>>>(pos, idx1, pos1);
  nbr_kernel<N1, M1><<<(BB * M1) / 2, 64, 0, stream>>>(pos, pos1, nbr1, 0.2f * 0.2f);
  sa_kernel<N1, M1, 3, 32, 32, 32, 64><<<(BB * M1) / 2, 64, 0, stream>>>(
      pos, nullptr, pos1, nbr1, wt10, s1b0, wt11, s1b1, wt12, s1b2, x1);

  // ---- SA2 ----
  fps_kernel<M1, M2><<<BB, 256, 0, stream>>>(pos1, idx2, pos2);
  nbr_kernel<M1, M2><<<(BB * M2) / 2, 64, 0, stream>>>(pos1, pos2, nbr2, 0.4f * 0.4f);
  sa_kernel<M1, M2, 67, 96, 64, 64, 128><<<(BB * M2) / 2, 64, 0, stream>>>(
      pos1, x1, pos2, nbr2, wt20, s2b0, wt21, s2b1, wt22, s2b2, x2);

  // ---- SA3 + global max pool ----
  sa3_kernel<<<(BB * 32) / 2, 64, 0, stream>>>(x2, pos2, wt30, s3b0, wt31, s3b1,
                                               wt32, s3b2, gpart);
  gmax_kernel<<<(BB * 512 + 255) / 256, 256, 0, stream>>>(gpart, feat);

  // ---- VAE head ----
  head_kernel<<<1, 256, 0, stream>>>(feat, wmu, bmu, wlv, blv, (float*)d_out);
}
